// QKVAttentionLegacy_34488587387357
// MI455X (gfx1250) — matrix operand order repaired and checked by reading, not compiled, not run on hardware
//
#include <hip/hip_runtime.h>

typedef __attribute__((ext_vector_type(16))) _Float16 v16h;
typedef __attribute__((ext_vector_type(8)))  float    v8f;

#define LEN   2048
#define NCH   64
#define SCH   32
#define WGQ   128   // 8 waves * 16 queries

// xor-16 lane swap within a wave32 (ds_swizzle: offset[14:10]=xor=0x10, and=0x1f)
__device__ __forceinline__ float xor16(float v) {
    int x = __builtin_bit_cast(int, v);
    x = __builtin_amdgcn_ds_swizzle(x, 0x401F);
    return __builtin_bit_cast(float, x);
}

__global__ __launch_bounds__(256)
void qkv_attn_f16_wmma(const float* __restrict__ qkv, float* __restrict__ out) {
    const int lane = threadIdx.x & 31;
    const int wid  = threadIdx.x >> 5;
    const int n    = lane & 15;   // column / M-within-tile index
    const int hf   = lane >> 4;   // which half of the wave

    const int head = blockIdx.y;           // 0..31 = b*8 + hb
    const int b    = head >> 3;
    const int hb   = head & 7;
    const int tq   = blockIdx.x * WGQ + wid * 16;  // this wave's query base

    const float* baseq = qkv + (size_t)(b * 1536 + hb * 192) * LEN;
    const float* basek = baseq + (size_t)NCH * LEN;
    const float* basev = baseq + (size_t)(2 * NCH) * LEN;
    float*       baseo = out + (size_t)(b * 512 + hb * NCH) * LEN;

    __shared__ _Float16 kl[SCH][NCH];   // [s][c] : A-operand (K^T) pairs contiguous in c
    __shared__ _Float16 vl[NCH][SCH];   // [c][s] : A-operand (V)  pairs contiguous in s

    // ---- Q as B-fragments, scale^2 = 1/sqrt(ch) = 1/8 folded in, held all loop ----
    // B layout (16-bit, 32x16): lane<16: elem e -> K = e + (e&8) ; lane>=16: K = e+8+(e&8) ; N = n
    v16h bq0, bq1;
#pragma unroll
    for (int e = 0; e < 16; ++e) {
        int c = e + (e & 8) + hf * 8;
        bq0[e] = (_Float16)(0.125f * baseq[(size_t)c        * LEN + tq + n]);
        bq1[e] = (_Float16)(0.125f * baseq[(size_t)(c + 32) * LEN + tq + n]);
    }

    v8f O[4];
#pragma unroll
    for (int cb = 0; cb < 4; ++cb) O[cb] = (v8f){};
    float m = -1e30f, l = 0.f;

    for (int s0 = 0; s0 < LEN; s0 += SCH) {
        __syncthreads();
        // ---- stage K/V chunk: coalesced f32 reads, cvt to f16 in LDS ----
#pragma unroll
        for (int i = 0; i < 8; ++i) {
            int ei = threadIdx.x + i * 256;   // 0..2047 over 64c x 32s
            int c = ei >> 5, s = ei & 31;
            kl[s][c] = (_Float16)basek[(size_t)c * LEN + s0 + s];
            vl[c][s] = (_Float16)basev[(size_t)c * LEN + s0 + s];
        }
        __syncthreads();
        if (s0 + SCH < LEN) {   // uniform branch: prefetch next chunk lines
            int c = threadIdx.x & 63;
            __builtin_prefetch(basek + (size_t)c * LEN + s0 + SCH, 0, 3);
            __builtin_prefetch(basev + (size_t)c * LEN + s0 + SCH, 0, 3);
        }

        // ---- St = K^T x Q  (two 16x16 s-tiles, K-dim c=64 split 32+32) ----
        // A layout (16-bit, 16x32): M = n ; lane<16: K = e + (e&8) ; lane>=16: K = e+8+(e&8)
        v16h a;
        v8f S0 = {}, S1 = {};
#pragma unroll
        for (int e = 0; e < 16; ++e) a[e] = kl[n][e + (e & 8) + hf * 8];
        S0 = __builtin_amdgcn_wmma_f32_16x16x32_f16(false, a, false, bq0, (short)0, S0, false, false);
#pragma unroll
        for (int e = 0; e < 16; ++e) a[e] = kl[n][32 + e + (e & 8) + hf * 8];
        S0 = __builtin_amdgcn_wmma_f32_16x16x32_f16(false, a, false, bq1, (short)0, S0, false, false);
#pragma unroll
        for (int e = 0; e < 16; ++e) a[e] = kl[16 + n][e + (e & 8) + hf * 8];
        S1 = __builtin_amdgcn_wmma_f32_16x16x32_f16(false, a, false, bq0, (short)0, S1, false, false);
#pragma unroll
        for (int e = 0; e < 16; ++e) a[e] = kl[16 + n][32 + e + (e & 8) + hf * 8];
        S1 = __builtin_amdgcn_wmma_f32_16x16x32_f16(false, a, false, bq1, (short)0, S1, false, false);

        // ---- online softmax over s (rows of St) : per-lane scalar state ----
        float mn = m;
#pragma unroll
        for (int r = 0; r < 8; ++r) mn = fmaxf(mn, fmaxf(S0[r], S1[r]));
        mn = fmaxf(mn, xor16(mn));
        float alpha = __expf(m - mn);
        m = mn;
        float rs = 0.f;
#pragma unroll
        for (int r = 0; r < 8; ++r) {
            float p0 = __expf(S0[r] - mn); S0[r] = p0;
            float p1 = __expf(S1[r] - mn); S1[r] = p1;
            rs += p0 + p1;
        }
        rs += xor16(rs);
        l = l * alpha + rs;
#pragma unroll
        for (int cb = 0; cb < 4; ++cb)
#pragma unroll
            for (int r = 0; r < 8; ++r) O[cb][r] *= alpha;

        // ---- P^T (C/D layout) -> B-fragment for V x P^T, via xor-16 only ----
        v16h bp;
#pragma unroll
        for (int e = 0; e < 8; ++e) {
            float x0 = S0[e], x1 = S1[e];
            float p0 = xor16(x0);        // partner half's tile0 row
            float p1 = xor16(x1);        // partner half's tile1 row
            bp[e]     = (_Float16)(hf ? p1 : x0);   // K = e (+16 for hi half)
            bp[e + 8] = (_Float16)(hf ? x1 : p0);   // K = e+8 (+16 for hi half)
        }

        // ---- O[c,t] += V(16c x 32s) x P^T(32s x 16t), 4 channel blocks ----
#pragma unroll
        for (int cb = 0; cb < 4; ++cb) {
            v16h av;
#pragma unroll
            for (int e = 0; e < 16; ++e) av[e] = vl[cb * 16 + n][e + hf * 16];
            O[cb] = __builtin_amdgcn_wmma_f32_16x16x32_f16(false, av, false, bp, (short)0, O[cb], false, false);
        }
    }

    // ---- normalize and store: row M=r(+8 for hi half) -> channel, N=n -> t ----
    float inv = 1.0f / l;
#pragma unroll
    for (int cb = 0; cb < 4; ++cb)
#pragma unroll
        for (int r = 0; r < 8; ++r) {
            int c = cb * 16 + r + hf * 8;
            baseo[(size_t)c * LEN + tq + n] = O[cb][r] * inv;
        }
}

extern "C" void kernel_launch(void* const* d_in, const int* in_sizes, int n_in,
                              void* d_out, int out_size, void* d_ws, size_t ws_size,
                              hipStream_t stream) {
    (void)in_sizes; (void)n_in; (void)out_size; (void)d_ws; (void)ws_size;
    const float* qkv = (const float*)d_in[0];
    float* out = (float*)d_out;
    dim3 grid(LEN / WGQ, 32);   // 16 query-tiles x 32 heads
    qkv_attn_f16_wmma<<<grid, 256, 0, stream>>>(qkv, out);
}
